// EnhancerGrammarTransformer_83210696393643
// MI455X (gfx1250) — matrix operand order repaired and checked by reading, not compiled, not run on hardware
//
#include <hip/hip_runtime.h>
#include <hip/hip_bf16.h>

// ---------------------------------------------------------------------------
// Model dims
// ---------------------------------------------------------------------------
#define BB   16
#define SS   512
#define DD   256
#define HH   8
#define DHH  32
#define GG   10
#define NM   512
#define FFN  2048
#define BIO  20
#define MROWS (BB*SS)          // 8192
#define EPSLN 1e-5f

typedef __attribute__((ext_vector_type(16))) __bf16 v16bf;
typedef __attribute__((ext_vector_type(8)))  __bf16 v8bf;
typedef __attribute__((ext_vector_type(4)))  __bf16 v4bf;
typedef __attribute__((ext_vector_type(8)))  float  v8f;

__device__ inline v8f wmma_bf16(v16bf a, v16bf b, v8f c) {
  return __builtin_amdgcn_wmma_f32_16x16x32_bf16(false, a, false, b, (short)0, c,
                                                 false, false);
}

__device__ inline v8bf ld8(const __bf16* p) { return *(const v8bf*)p; }

__device__ inline v16bf combine(v8bf lo, v8bf hi) {
  return __builtin_shufflevector(lo, hi, 0, 1, 2, 3, 4, 5, 6, 7,
                                 8, 9, 10, 11, 12, 13, 14, 15);
}

// fp32 -> bf16 vectorized conversion (n must be a multiple of 1024)
__global__ __launch_bounds__(256)
void cvt_bf16(const float* __restrict__ src, __bf16* __restrict__ dst) {
  int i = (blockIdx.x * 256 + threadIdx.x) * 4;
  float4 f = *(const float4*)(src + i);
  v4bf o;
  o[0] = (__bf16)f.x; o[1] = (__bf16)f.y; o[2] = (__bf16)f.z; o[3] = (__bf16)f.w;
  *(v4bf*)(dst + i) = o;
}

// ---------------------------------------------------------------------------
// Tiled GEMM: C[M,N] = act(A[M,K]bf16 @ W[N,K]bf16^T + bias[N])
// Block: 256 threads (8 waves), tile 128(M) x 64(N), K-step 32.
// Double-buffered LDS staging with global_load_async_to_lds_b128:
//   prologue loads tile 0; iteration i issues tile i+1 into the other buffer,
//   then s_wait_asynccnt 3 (in-order async completion => tile i resident)
//   overlaps tile i+1's DMA with tile i's 4 WMMAs.
// Wave (mw,nw) in 4x2 grid computes 32x32 (2x2 WMMA accumulators).
// ---------------------------------------------------------------------------
#define MT 128
#define NT 64
__global__ __launch_bounds__(256)
void gemm_wmma(const __bf16* __restrict__ A, const __bf16* __restrict__ Wm,
               const float* __restrict__ bias, float* __restrict__ C,
               int M, int N, int K, int act) {
  __shared__ __attribute__((aligned(16))) __bf16 sA[2][MT * 32];
  __shared__ __attribute__((aligned(16))) __bf16 sB[2][NT * 32];
  const int tid  = threadIdx.x;
  const int lane = tid & 31;
  const int wv   = tid >> 5;
  const int half = lane >> 4;
  const int mn   = lane & 15;
  const int mw   = wv >> 1;        // 0..3
  const int nw   = wv & 1;         // 0..1
  const int tn   = blockIdx.x, tm = blockIdx.y;

  // staging assignment: chunk row/segment for this thread (16B = 8 bf16)
  const int srow = tid >> 2;              // 0..63
  const int sseg = (tid & 3) * 8;         // 0,8,16,24

  unsigned la0[2], la1[2], lb[2];
#pragma unroll
  for (int bf = 0; bf < 2; ++bf) {
    la0[bf] = (unsigned)(uintptr_t)&sA[bf][srow * 32 + sseg];
    la1[bf] = (unsigned)(uintptr_t)&sA[bf][(64 + srow) * 32 + sseg];
    lb[bf]  = (unsigned)(uintptr_t)&sB[bf][srow * 32 + sseg];
  }

  const __bf16* gA0 = A  + (size_t)(tm * MT +      srow) * K + sseg;
  const __bf16* gA1 = A  + (size_t)(tm * MT + 64 + srow) * K + sseg;
  const __bf16* gB  = Wm + (size_t)(tn * NT +      srow) * K + sseg;

  auto issue = [&](int k0, int bf) {
    asm volatile("global_load_async_to_lds_b128 %0, %1, off"
                 :: "v"(la0[bf]), "v"((unsigned long long)(uintptr_t)(gA0 + k0)) : "memory");
    asm volatile("global_load_async_to_lds_b128 %0, %1, off"
                 :: "v"(la1[bf]), "v"((unsigned long long)(uintptr_t)(gA1 + k0)) : "memory");
    asm volatile("global_load_async_to_lds_b128 %0, %1, off"
                 :: "v"(lb[bf]),  "v"((unsigned long long)(uintptr_t)(gB  + k0)) : "memory");
  };

  v8f acc[2][2] = {};
  const int nk = K / 32;

  issue(0, 0);   // prologue: stage tile 0

  for (int i = 0; i < nk; ++i) {
    const int cur = i & 1;
    if (i + 1 < nk) {
      issue((i + 1) * 32, cur ^ 1);                 // prefetch next tile
      asm volatile("s_wait_asynccnt 3" ::: "memory"); // tile i resident
    } else {
      asm volatile("s_wait_asynccnt 0" ::: "memory");
    }
    __syncthreads();   // tile i visible to all waves

    v16bf af[2], bfr[2];
#pragma unroll
    for (int ms = 0; ms < 2; ++ms) {
      int row = mw * 32 + ms * 16 + mn;
      af[ms] = combine(ld8(&sA[cur][row * 32 + half * 8]),
                       ld8(&sA[cur][row * 32 + 16 + half * 8]));
    }
#pragma unroll
    for (int ns = 0; ns < 2; ++ns) {
      int rw = nw * 32 + ns * 16 + mn;
      bfr[ns] = combine(ld8(&sB[cur][rw * 32 + half * 8]),
                        ld8(&sB[cur][rw * 32 + 16 + half * 8]));
    }
#pragma unroll
    for (int ms = 0; ms < 2; ++ms)
#pragma unroll
      for (int ns = 0; ns < 2; ++ns)
        acc[ms][ns] = wmma_bf16(af[ms], bfr[ns], acc[ms][ns]);

    __syncthreads();   // all waves done reading buffer `cur` before reuse
  }

#pragma unroll
  for (int ns = 0; ns < 2; ++ns) {
    const int   n  = tn * NT + nw * 32 + ns * 16 + mn;
    const float bv = bias ? bias[n] : 0.0f;
#pragma unroll
    for (int ms = 0; ms < 2; ++ms) {
#pragma unroll
      for (int v = 0; v < 8; ++v) {
        int   m   = tm * MT + mw * 32 + ms * 16 + v + half * 8;
        float val = acc[ms][ns][v] + bv;
        if (act == 1) val = fmaxf(val, 0.0f);
        C[(size_t)m * N + n] = val;
      }
    }
  }
}

// ---------------------------------------------------------------------------
// Fused attention (bf16 q/k/v): one wave per (q-tile, head, batch).
// 16x512 fp32 scores in LDS -> all-lane softmax -> bf16 probs in LDS -> P@V.
// ---------------------------------------------------------------------------
__global__ __launch_bounds__(32)
void attn_kernel(const __bf16* __restrict__ q, const __bf16* __restrict__ k,
                 const __bf16* __restrict__ v, float* __restrict__ o) {
  __shared__ float sc[16 * SS];
  __shared__ __attribute__((aligned(16))) __bf16 sp[16 * SS];
  __shared__ float red32[32];
  const int lane = threadIdx.x;
  const int half = lane >> 4;
  const int mn   = lane & 15;
  const int qt = blockIdx.x, h = blockIdx.y, b = blockIdx.z;

  const __bf16* qr = q + ((size_t)(b * SS + qt * 16 + mn) * DD) + h * DHH;
  v16bf af = combine(ld8(qr + half * 8), ld8(qr + 16 + half * 8));

  const float scale = 0.1767766952966369f; // 1/sqrt(32)
  for (int kt = 0; kt < SS / 16; ++kt) {
    const __bf16* kr = k + ((size_t)(b * SS + kt * 16 + mn) * DD) + h * DHH;
    v16bf bf_ = combine(ld8(kr + half * 8), ld8(kr + 16 + half * 8));
    v8f c = {};
    c = wmma_bf16(af, bf_, c);
#pragma unroll
    for (int vv = 0; vv < 8; ++vv)
      sc[(vv + half * 8) * SS + kt * 16 + mn] = c[vv] * scale;
  }
  __syncthreads();

  // softmax: lane (mn, half) owns row mn, columns [half*256, half*256+256)
  {
    float* row = sc + mn * SS;
    const int c0 = half * 256;
    float mx = -1e30f;
    for (int i = c0; i < c0 + 256; ++i) mx = fmaxf(mx, row[i]);
    red32[lane] = mx;
    __syncthreads();
    float gmx = fmaxf(red32[mn], red32[16 + mn]);
    __syncthreads();
    float s = 0.0f;
    for (int i = c0; i < c0 + 256; ++i) {
      float e = __expf(row[i] - gmx);
      row[i] = e;
      s += e;
    }
    red32[lane] = s;
    __syncthreads();
    float inv = 1.0f / (red32[mn] + red32[16 + mn]);
    __bf16* prow = sp + mn * SS;
    for (int i = c0; i < c0 + 256; ++i) prow[i] = (__bf16)(row[i] * inv);
  }
  __syncthreads();

  for (int nt = 0; nt < 2; ++nt) {
    v8f c = {};
    for (int kc = 0; kc < SS / 32; ++kc) {
      const __bf16* pr = sp + mn * SS + kc * 32;
      v16bf pa = combine(ld8(pr + half * 16), ld8(pr + half * 16 + 8));
      v16bf vb;
      const __bf16* vr = v + ((size_t)(b * SS + kc * 32 + half * 16) * DD) + h * DHH + nt * 16 + mn;
#pragma unroll
      for (int j = 0; j < 16; ++j) vb[j] = vr[(size_t)j * DD];
      c = wmma_bf16(pa, vb, c);
    }
    float* orow = o + ((size_t)(b * SS + qt * 16) * DD) + h * DHH + nt * 16 + mn;
#pragma unroll
    for (int vv = 0; vv < 8; ++vv) orow[(size_t)(vv + half * 8) * DD] = c[vv];
  }
}

// ---------------------------------------------------------------------------
// Embed: x = emb[ids] + bio @ bio_w^T + bio_b + pe[positions]
// ---------------------------------------------------------------------------
__global__ __launch_bounds__(256)
void embed_kernel(const int* __restrict__ ids, const float* __restrict__ bio,
                  const int* __restrict__ pos, const float* __restrict__ emb,
                  const float* __restrict__ bio_w, const float* __restrict__ bio_b,
                  const float* __restrict__ pe, float* __restrict__ x) {
  int bs = blockIdx.x, d = threadIdx.x;
  int s = bs & (SS - 1);
  int id = ids[bs];
  float acc = emb[(size_t)id * DD + d] + bio_b[d];
  const float* br = bio + (size_t)bs * BIO;
  const float* wr = bio_w + (size_t)d * BIO;
#pragma unroll
  for (int t = 0; t < BIO; ++t) acc += br[t] * wr[t];
  acc += pe[(size_t)pos[s] * DD + d];
  x[(size_t)bs * DD + d] = acc;
}

// k_in selection on bf16 (5 = flip, 4 = roll(+1), else identity)
__global__ __launch_bounds__(256)
void kin_bf16(const __bf16* __restrict__ x, const int* __restrict__ gidx,
              __bf16* __restrict__ kin) {
  int idx = blockIdx.x * 256 + threadIdx.x;
  int d = idx & (DD - 1);
  int s = (idx >> 8) & (SS - 1);
  int b = idx >> 17;
  int g = gidx[0];
  int src = (g == 5) ? (SS - 1 - s) : ((g == 4) ? ((s + SS - 1) & (SS - 1)) : s);
  kin[idx] = x[(((size_t)(b * SS + src)) << 8) + d];
}

// ---------------------------------------------------------------------------
// Sparse-gated edge aggregation: agg[b,i,:] = sum_j A[i,j]*E[i,j,:]*x[b,j,:]
// ---------------------------------------------------------------------------
__global__ __launch_bounds__(256)
void graph_agg(const float* __restrict__ adj, const float* __restrict__ E,
               const float* __restrict__ x, float* __restrict__ agg) {
  int i = blockIdx.x, d = threadIdx.x;
  float acc[BB];
#pragma unroll
  for (int b = 0; b < BB; ++b) acc[b] = 0.0f;
  for (int j = 0; j < NM; ++j) {
    float a = adj[i * NM + j];
    if (a != 0.0f) {
      float e = a * E[((size_t)(i * NM + j)) * DD + d];
#pragma unroll
      for (int b = 0; b < BB; ++b)
        acc[b] += e * x[(((size_t)(b * SS + j)) << 8) + d];
    }
  }
#pragma unroll
  for (int b = 0; b < BB; ++b)
    agg[(((size_t)(b * SS + i)) << 8) + d] = acc[b];
}

__device__ inline float sigmf(float x) { return 1.0f / (1.0f + __expf(-x)); }

__global__ __launch_bounds__(256)
void gru_kernel(const float* __restrict__ gi, const float* __restrict__ gh,
                const float* __restrict__ x, float* __restrict__ out) {
  int idx = blockIdx.x * 256 + threadIdx.x;
  int d = idx & (DD - 1);
  int row = idx >> 8;
  size_t base = (size_t)row * (3 * DD) + d;
  float ir = gi[base], iz = gi[base + DD], inn = gi[base + 2 * DD];
  float hr = gh[base], hz = gh[base + DD], hn  = gh[base + 2 * DD];
  float r = sigmf(ir + hr);
  float z = sigmf(iz + hz);
  float n = tanhf(inn + r * hn);
  out[idx] = (1.0f - z) * n + z * x[idx];
}

// ---------------------------------------------------------------------------
// Dynamic conv
// ---------------------------------------------------------------------------
__global__ __launch_bounds__(256)
void conv_kern_kernel(const float* __restrict__ attn, const float* __restrict__ cw,
                      const float* __restrict__ cb, float* __restrict__ kern) {
  __shared__ float md[DD];
  int b = blockIdx.x, d = threadIdx.x;
  float s = 0.0f;
  for (int j = 0; j < SS; ++j) s += attn[(((size_t)(b * SS + j)) << 8) + d];
  md[d] = s * (1.0f / SS);
  __syncthreads();
  if (d < 3) {
    float acc = cb[d];
    for (int t = 0; t < DD; ++t) acc += md[t] * cw[d * DD + t];
    kern[b * 3 + d] = acc;
  }
}

__global__ __launch_bounds__(256)
void conv_apply(const float* __restrict__ x, const float* __restrict__ kern,
                float* __restrict__ out) {
  int idx = blockIdx.x * 256 + threadIdx.x;
  int s = (idx >> 8) & (SS - 1);
  int b = idx >> 17;
  float k0 = kern[b * 3 + 0], k1 = kern[b * 3 + 1], k2 = kern[b * 3 + 2];
  float xm1 = (s > 0)      ? x[idx - DD] : 0.0f;
  float xp1 = (s < SS - 1) ? x[idx + DD] : 0.0f;
  out[idx] = k0 * xm1 + k1 * x[idx] + k2 * xp1;
}

// ---------------------------------------------------------------------------
// out = LayerNorm(a + r) * g + beta
// ---------------------------------------------------------------------------
__global__ __launch_bounds__(256)
void add_ln(const float* __restrict__ a, const float* __restrict__ r,
            const float* __restrict__ g, const float* __restrict__ be,
            float* __restrict__ out) {
  __shared__ float red[256];
  int row = blockIdx.x, d = threadIdx.x;
  size_t idx = (size_t)row * DD + d;
  float v = a[idx] + r[idx];
  red[d] = v;
  __syncthreads();
  for (int st = 128; st > 0; st >>= 1) { if (d < st) red[d] += red[d + st]; __syncthreads(); }
  float m = red[0] * (1.0f / DD);
  __syncthreads();
  float c = v - m;
  red[d] = c * c;
  __syncthreads();
  for (int st = 128; st > 0; st >>= 1) { if (d < st) red[d] += red[d + st]; __syncthreads(); }
  float var = red[0] * (1.0f / DD);
  out[idx] = c * rsqrtf(var + EPSLN) * g[d] + be[d];
}

// ---------------------------------------------------------------------------
// Heads
// ---------------------------------------------------------------------------
__global__ __launch_bounds__(32)
void syntax_kernel(const float* __restrict__ h, const float* __restrict__ w,
                   const float* __restrict__ bi, float* __restrict__ out) {
  __shared__ float lg[GG];
  int row = blockIdx.x, lane = threadIdx.x;
  if (lane < GG) {
    float acc = bi[lane];
    const float* hr = h + (size_t)row * DD;
    const float* wr = w + lane * DD;
    for (int t = 0; t < DD; ++t) acc += hr[t] * wr[t];
    lg[lane] = acc;
  }
  __syncthreads();
  if (lane < GG) {
    float mx = lg[0];
#pragma unroll
    for (int t = 1; t < GG; ++t) mx = fmaxf(mx, lg[t]);
    float s = 0.0f;
#pragma unroll
    for (int t = 0; t < GG; ++t) s += __expf(lg[t] - mx);
    out[(size_t)row * GG + lane] = __expf(lg[lane] - mx) / s;
  }
}

__global__ __launch_bounds__(256)
void mean_kernel(const float* __restrict__ x, float* __restrict__ xm) {
  int b = blockIdx.x, d = threadIdx.x;
  float s = 0.0f;
  for (int j = 0; j < SS; ++j) s += x[(((size_t)(b * SS + j)) << 8) + d];
  xm[b * DD + d] = s * (1.0f / SS);
}

__global__ __launch_bounds__(128)
void disc_kernel(const float* __restrict__ xm, const float* __restrict__ w1,
                 const float* __restrict__ b1, const float* __restrict__ w2,
                 const float* __restrict__ b2, float* __restrict__ out) {
  __shared__ float hid[DD / 2];
  int b = blockIdx.x, t = threadIdx.x;
  float acc = b1[t];
  for (int i = 0; i < DD; ++i) acc += xm[b * DD + i] * w1[t * DD + i];
  hid[t] = fmaxf(acc, 0.0f);
  __syncthreads();
  if (t == 0) {
    float s = b2[0];
    for (int i = 0; i < DD / 2; ++i) s += hid[i] * w2[i];
    out[b] = sigmf(s);
  }
}

__global__ __launch_bounds__(32)
void logits_kernel(const float* __restrict__ xm, const float* __restrict__ w,
                   const float* __restrict__ bi, float* __restrict__ out) {
  int b = blockIdx.x, t = threadIdx.x;
  if (t < GG) {
    float acc = bi[t];
    for (int i = 0; i < DD; ++i) acc += xm[b * DD + i] * w[t * DD + i];
    out[b * GG + t] = acc;
  }
}

__global__ __launch_bounds__(256)
void latent_kernel(const float* __restrict__ xm, const float* __restrict__ w,
                   const float* __restrict__ bi, float* __restrict__ out) {
  int b = blockIdx.x, d = threadIdx.x;
  float acc = bi[d];
  for (int i = 0; i < DD; ++i) acc += xm[b * DD + i] * w[d * DD + i];
  out[b * DD + d] = acc;
}

// ---------------------------------------------------------------------------
// Host-side orchestration
// ---------------------------------------------------------------------------
static inline void gemm(hipStream_t s, const __bf16* A, const __bf16* W,
                        const float* bias, float* C, int M, int N, int K, int act) {
  dim3 g(N / NT, M / MT);
  gemm_wmma<<<g, 256, 0, s>>>(A, W, bias, C, M, N, K, act);
}

static inline void cvt(hipStream_t s, const float* src, __bf16* dst, int n) {
  cvt_bf16<<<n / 1024, 256, 0, s>>>(src, dst);
}

extern "C" void kernel_launch(void* const* d_in, const int* in_sizes, int n_in,
                              void* d_out, int out_size, void* d_ws, size_t ws_size,
                              hipStream_t stream) {
  (void)in_sizes; (void)n_in; (void)out_size; (void)ws_size;
  const int*   motif_ids = (const int*)d_in[0];
  const float* bio       = (const float*)d_in[1];
  const int*   positions = (const int*)d_in[2];
  const float* adj       = (const float*)d_in[3];
  const int*   gidx      = (const int*)d_in[4];
  const float* emb   = (const float*)d_in[5];
  const float* bio_w = (const float*)d_in[6];
  const float* bio_b = (const float*)d_in[7];
  const float* pe    = (const float*)d_in[8];
  const float* ga_wi = (const float*)d_in[9];
  const float* ga_bi = (const float*)d_in[10];
  const float* ga_wo = (const float*)d_in[11];
  const float* ga_bo = (const float*)d_in[12];
  const float* edge  = (const float*)d_in[13];
  const float* gru_wih = (const float*)d_in[14];
  const float* gru_whh = (const float*)d_in[15];
  const float* gru_bih = (const float*)d_in[16];
  const float* gru_bhh = (const float*)d_in[17];
  const float* conv_w  = (const float*)d_in[18];
  const float* conv_b  = (const float*)d_in[19];
  const float* rt_wi  = (const float*)d_in[20];
  const float* rt_bi  = (const float*)d_in[21];
  const float* rt_wo  = (const float*)d_in[22];
  const float* rt_bo  = (const float*)d_in[23];
  const float* rt_l1w = (const float*)d_in[24];
  const float* rt_l1b = (const float*)d_in[25];
  const float* rt_l2w = (const float*)d_in[26];
  const float* rt_l2b = (const float*)d_in[27];
  const float* rt_g1  = (const float*)d_in[28];
  const float* rt_b1  = (const float*)d_in[29];
  const float* rt_g2  = (const float*)d_in[30];
  const float* rt_b2  = (const float*)d_in[31];
  const float* syn_w  = (const float*)d_in[32];
  const float* syn_b  = (const float*)d_in[33];
  const float* d1w = (const float*)d_in[34];
  const float* d1b = (const float*)d_in[35];
  const float* d2w = (const float*)d_in[36];
  const float* d2b = (const float*)d_in[37];
  const float* cls_w = (const float*)d_in[38];
  const float* cls_b = (const float*)d_in[39];
  const float* lat_w = (const float*)d_in[40];
  const float* lat_b = (const float*)d_in[41];

  float* out = (float*)d_out;
  const size_t TBSD = (size_t)BB * SS * DD;             // 2,097,152
  float* o_attn   = out;
  float* o_graph  = out + TBSD;
  float* o_conv   = out + 2 * TBSD;
  float* o_h      = out + 3 * TBSD;
  float* o_syntax = out + 4 * TBSD;
  float* o_disc   = o_syntax + (size_t)BB * SS * GG;
  float* o_logits = o_disc + BB;
  float* o_latent = o_logits + (size_t)BB * GG;

  // ---- fp32 workspace ----
  float* W = (float*)d_ws;
  float* w_x   = W;
  float* w_q   = W + 1 * TBSD;
  float* w_k   = W + 2 * TBSD;
  float* w_v   = W + 3 * TBSD;
  float* w_ah  = W + 4 * TBSD;
  float* w_h   = W + 5 * TBSD;
  float* w_big = W + 6 * TBSD;          // graph: gi/gh/agg ; rt: ff1 (8*TBSD)
  float* w_gi  = w_big;                 // [M,3D]
  float* w_gh  = w_big + 3 * TBSD;      // [M,3D]
  float* w_agg = w_big + 6 * TBSD;      // [B,S,D]
  float* w_ff1 = w_big;                 // [M,FF] (reuse after graph phase)
  float* w_small = W + 14 * TBSD;
  float* w_kern = w_small;              // [B,3]
  float* w_xm   = w_small + 64;         // [B,D]

  // ---- bf16 workspace ----
  __bf16* BW = (__bf16*)(W + 14 * TBSD + 8192);
  __bf16* b_x   = BW;
  __bf16* b_kin = BW + 1 * TBSD;
  __bf16* b_q   = BW + 2 * TBSD;
  __bf16* b_k   = BW + 3 * TBSD;
  __bf16* b_v   = BW + 4 * TBSD;
  __bf16* b_ah  = BW + 5 * TBSD;
  __bf16* b_agg = BW + 6 * TBSD;
  __bf16* b_h   = BW + 7 * TBSD;
  __bf16* b_h2  = BW + 8 * TBSD;
  __bf16* b_ff1 = BW + 9 * TBSD;        // M*FFN = 8*TBSD
  __bf16* c_w   = BW + 17 * TBSD;       // converted weights
  __bf16* c_ga_wi   = c_w;                       // 3*D*D
  __bf16* c_ga_wo   = c_ga_wi + 3 * DD * DD;     // D*D
  __bf16* c_gru_wih = c_ga_wo + DD * DD;         // 3*D*D
  __bf16* c_gru_whh = c_gru_wih + 3 * DD * DD;   // 3*D*D
  __bf16* c_rt_wi   = c_gru_whh + 3 * DD * DD;   // 3*D*D
  __bf16* c_rt_wo   = c_rt_wi + 3 * DD * DD;     // D*D
  __bf16* c_rt_l1w  = c_rt_wo + DD * DD;         // FF*D
  __bf16* c_rt_l2w  = c_rt_l1w + FFN * DD;       // D*FF

  const int nBSD = (int)(TBSD / 256);

  // 0. one-time weight conversions (deterministic every call)
  cvt(stream, ga_wi,   c_ga_wi,   3 * DD * DD);
  cvt(stream, ga_wo,   c_ga_wo,   DD * DD);
  cvt(stream, gru_wih, c_gru_wih, 3 * DD * DD);
  cvt(stream, gru_whh, c_gru_whh, 3 * DD * DD);
  cvt(stream, rt_wi,   c_rt_wi,   3 * DD * DD);
  cvt(stream, rt_wo,   c_rt_wo,   DD * DD);
  cvt(stream, rt_l1w,  c_rt_l1w,  FFN * DD);
  cvt(stream, rt_l2w,  c_rt_l2w,  DD * FFN);

  // 1. embedding + bf16 copy
  embed_kernel<<<MROWS, 256, 0, stream>>>(motif_ids, bio, positions, emb,
                                          bio_w, bio_b, pe, w_x);
  cvt(stream, w_x, b_x, (int)TBSD);
  kin_bf16<<<nBSD, 256, 0, stream>>>(b_x, gidx, b_kin);

  // 2. grammar attention -> o_attn
  gemm(stream, b_x,   c_ga_wi,               ga_bi,          w_q, MROWS, DD, DD, 0);
  gemm(stream, b_kin, c_ga_wi + DD * DD,     ga_bi + DD,     w_k, MROWS, DD, DD, 0);
  gemm(stream, b_x,   c_ga_wi + 2 * DD * DD, ga_bi + 2 * DD, w_v, MROWS, DD, DD, 0);
  cvt(stream, w_q, b_q, (int)TBSD);
  cvt(stream, w_k, b_k, (int)TBSD);
  cvt(stream, w_v, b_v, (int)TBSD);
  attn_kernel<<<dim3(SS / 16, HH, BB), 32, 0, stream>>>(b_q, b_k, b_v, w_ah);
  cvt(stream, w_ah, b_ah, (int)TBSD);
  gemm(stream, b_ah, c_ga_wo, ga_bo, o_attn, MROWS, DD, DD, 0);

  // 3. motif interaction graph + GRU -> o_graph
  graph_agg<<<NM, 256, 0, stream>>>(adj, edge, w_x, w_agg);
  cvt(stream, w_agg, b_agg, (int)TBSD);
  gemm(stream, b_agg, c_gru_wih, gru_bih, w_gi, MROWS, 3 * DD, DD, 0);
  gemm(stream, b_x,   c_gru_whh, gru_bhh, w_gh, MROWS, 3 * DD, DD, 0);
  gru_kernel<<<nBSD, 256, 0, stream>>>(w_gi, w_gh, w_x, o_graph);

  // 4. attention-guided dynamic conv -> o_conv
  conv_kern_kernel<<<BB, 256, 0, stream>>>(o_attn, conv_w, conv_b, w_kern);
  conv_apply<<<nBSD, 256, 0, stream>>>(w_x, w_kern, o_conv);

  // 5. recursive transformer block, depth 3
  hipMemcpyAsync(w_h, w_x, TBSD * sizeof(float), hipMemcpyDeviceToDevice, stream);
  hipMemcpyAsync(b_h, b_x, TBSD * sizeof(__bf16), hipMemcpyDeviceToDevice, stream);
  for (int it = 0; it < 3; ++it) {
    gemm(stream, b_h, c_rt_wi,               rt_bi,          w_q, MROWS, DD, DD, 0);
    gemm(stream, b_h, c_rt_wi + DD * DD,     rt_bi + DD,     w_k, MROWS, DD, DD, 0);
    gemm(stream, b_h, c_rt_wi + 2 * DD * DD, rt_bi + 2 * DD, w_v, MROWS, DD, DD, 0);
    cvt(stream, w_q, b_q, (int)TBSD);
    cvt(stream, w_k, b_k, (int)TBSD);
    cvt(stream, w_v, b_v, (int)TBSD);
    attn_kernel<<<dim3(SS / 16, HH, BB), 32, 0, stream>>>(b_q, b_k, b_v, w_ah);
    cvt(stream, w_ah, b_ah, (int)TBSD);
    gemm(stream, b_ah, c_rt_wo, rt_bo, w_q /*tmp*/, MROWS, DD, DD, 0);
    add_ln<<<MROWS, 256, 0, stream>>>(w_h, w_q, rt_g1, rt_b1, w_k /*h2*/);
    cvt(stream, w_k, b_h2, (int)TBSD);
    gemm(stream, b_h2, c_rt_l1w, rt_l1b, w_ff1, MROWS, FFN, DD, 1 /*relu*/);
    cvt(stream, w_ff1, b_ff1, (int)(8 * TBSD));
    gemm(stream, b_ff1, c_rt_l2w, rt_l2b, w_v /*ff2*/, MROWS, DD, FFN, 0);
    add_ln<<<MROWS, 256, 0, stream>>>(w_k, w_v, rt_g2, rt_b2, w_h);
    cvt(stream, w_h, b_h, (int)TBSD);
  }
  hipMemcpyAsync(o_h, w_h, TBSD * sizeof(float), hipMemcpyDeviceToDevice, stream);

  // 6. heads
  syntax_kernel<<<MROWS, 32, 0, stream>>>(w_h, syn_w, syn_b, o_syntax);
  mean_kernel<<<BB, 256, 0, stream>>>(w_x, w_xm);
  disc_kernel<<<BB, 128, 0, stream>>>(w_xm, d1w, d1b, d2w, d2b, o_disc);
  logits_kernel<<<BB, 32, 0, stream>>>(w_xm, cls_w, cls_b, o_logits);
  latent_kernel<<<BB, 256, 0, stream>>>(w_xm, lat_w, lat_b, o_latent);
}
